// HigherOrderAttention_33097017983134
// MI455X (gfx1250) — matrix operand order repaired and checked by reading, not compiled, not run on hardware
//
#include <hip/hip_runtime.h>
#include <hip/hip_bf16.h>
#include <math.h>

// ---------------------------------------------------------------------------
// Higher-order (order-3) attention for MI455X (gfx1250, wave32, WMMA).
// All matmuls run through v_wmma_f32_16x16x32_bf16.
// Round 2: k-outer loop with 4 persistent accumulators per wave (A operand
// loaded once per k-step, reused across 4 column tiles) + global prefetch.
// ---------------------------------------------------------------------------

typedef __attribute__((ext_vector_type(16))) __bf16 v16bf;
typedef __attribute__((ext_vector_type(8)))  __bf16 v8bf;
typedef __attribute__((ext_vector_type(8)))  float  v8f;

#define NH     8
#define HDIM   64
#define EMB    512
#define BB     2
#define TSEQ   512
#define MROWS  (BB*TSEQ)   // 1024
#define KKEEP  64

// ---- WMMA operand loaders (wave32) ----------------------------------------
// A operand (16x32 bf16): lane l: row m = row0 + (l&15), hi = l>>4.
//   elems 0..7  -> K = k0 + 8*hi + i          (contiguous 16B)
//   elems 8..15 -> K = k0 + 16 + 8*hi + (i-8) (contiguous 16B)
static __device__ __forceinline__ v16bf load_a(const __bf16* base, int ld, int row0, int k0) {
  int l  = threadIdx.x & 31;
  int m  = row0 + (l & 15);
  int hi = l >> 4;
  const __bf16* p = base + (size_t)m * ld + k0 + hi * 8;
  v8bf lo = *(const v8bf*)(p);
  v8bf hv = *(const v8bf*)(p + 16);
  return __builtin_shufflevector(lo, hv, 0,1,2,3,4,5,6,7,8,9,10,11,12,13,14,15);
}

// B operand (32x16 bf16), fed from a row-major "B^T" (N x K) buffer:
// lane l: col n = n0 + (l&15), hi = l>>4; elems 0..7 -> K = k0 + 8*hi + i, elems 8..15 -> K = k0 + 16 + 8*hi + (i-8)
static __device__ __forceinline__ v16bf load_b(const __bf16* base, int ld, int n0, int k0) {
  int l  = threadIdx.x & 31;
  int n  = n0 + (l & 15);
  int hi = l >> 4;
  const __bf16* p = base + (size_t)n * ld + k0 + hi * 8;
  v8bf lo = *(const v8bf*)(p);
  v8bf hv = *(const v8bf*)(p + 16);
  return __builtin_shufflevector(lo, hv, 0,1,2,3,4,5,6,7,8,9,10,11,12,13,14,15);
}

// ---- fp32 -> bf16 convert (optionally transposed into N x K) --------------
__global__ void ho_convert_bf16(const float* __restrict__ src, __bf16* __restrict__ dst,
                                int rows, int cols, int transpose) {
  int i = blockIdx.x * blockDim.x + threadIdx.x;
  if (i >= rows * cols) return;
  int r = i / cols, c = i - r * cols;
  float v = src[i];
  if (transpose) dst[(size_t)c * rows + r] = (__bf16)v;
  else           dst[i] = (__bf16)v;
}

// ---- generic bf16 WMMA GEMM: C(MxN,f32) = A(MxK) * BT(NxK)^T --------------
// Block = 4 waves -> 64x64 C tile; each wave owns one 16-row strip and all 4
// column tiles (4 accumulators), so the A operand is loaded once per k-step.
__global__ __launch_bounds__(128) void ho_gemm_bf16(const __bf16* __restrict__ A,
                                                    const __bf16* __restrict__ BT,
                                                    float* __restrict__ C,
                                                    int M, int N, int K) {
  int wave = threadIdx.x >> 5;
  int l    = threadIdx.x & 31;
  int hi   = l >> 4, n = l & 15;
  int row0 = blockIdx.x * 64 + wave * 16;
  int bcol = blockIdx.y * 64;

  v8f acc[4] = {{}, {}, {}, {}};
  for (int k0 = 0; k0 < K; k0 += 32) {
    if (k0 + 32 < K) {  // prefetch next A chunk (-> global_prefetch_b8)
      __builtin_prefetch(A + (size_t)(row0 + (l & 15)) * K + k0 + 32, 0, 3);
      __builtin_prefetch(BT + (size_t)(bcol + l) * K + k0 + 32, 0, 3);
    }
    v16bf a = load_a(A, K, row0, k0);
#pragma unroll
    for (int tj = 0; tj < 4; ++tj) {
      v16bf b = load_b(BT, K, bcol + tj * 16, k0);
      acc[tj] = __builtin_amdgcn_wmma_f32_16x16x32_bf16(false, a, false, b,
                                                        (short)0, acc[tj], false, false);
    }
  }
#pragma unroll
  for (int tj = 0; tj < 4; ++tj) {
    int col0 = bcol + tj * 16;
#pragma unroll
    for (int r = 0; r < 8; ++r)
      C[(size_t)(row0 + r + 8 * hi) * N + col0 + n] = acc[tj][r];
  }
}

// ---- causal top-64 selection: one wave per (r,b,h,t) row -------------------
__global__ __launch_bounds__(128) void ho_topk(const float* __restrict__ Q,
                                               const float* __restrict__ K1,
                                               const float* __restrict__ K2,
                                               int* __restrict__ idxbuf) {
  __shared__ float sq[4][HDIM];
  int w   = threadIdx.x >> 5;
  int l   = threadIdx.x & 31;
  int wid = blockIdx.x * 4 + w;
  int t = wid % TSEQ;
  int h = (wid / TSEQ) % NH;
  int b = wid / (TSEQ * NH);
  int r = blockIdx.y;
  const float* Kr = (r == 0) ? K1 : K2;

  const float* qp = Q + (size_t)(b * TSEQ + t) * EMB + h * HDIM;
  sq[w][l]      = qp[l];
  sq[w][l + 32] = qp[l + 32];
  __syncthreads();

  float lv[16];
  for (int slot = 0; slot < 16; ++slot) {
    int s = slot * 32 + l;
    float acc = -INFINITY;
    if (s <= t) {
      const float* kp = Kr + (size_t)(b * TSEQ + s) * EMB + h * HDIM;
      acc = 0.0f;
      for (int d = 0; d < HDIM; ++d) acc += sq[w][d] * kp[d];
    }
    lv[slot] = acc;
  }

  int last_idx = 0;
  size_t obase = ((((size_t)r * BB + b) * NH + h) * TSEQ + t) * KKEEP;
  for (int rr = 0; rr < KKEEP; ++rr) {
    float bmax = -INFINITY; int bidx = 0x7fffffff;
    for (int slot = 0; slot < 16; ++slot) {
      int s = slot * 32 + l;
      float v = lv[slot];
      if (v > bmax || (v == bmax && s < bidx)) { bmax = v; bidx = s; }
    }
    for (int off = 16; off > 0; off >>= 1) {
      float ov = __shfl_xor(bmax, off, 32);
      int   oi = __shfl_xor(bidx, off, 32);
      if (ov > bmax || (ov == bmax && oi < bidx)) { bmax = ov; bidx = oi; }
    }
    if ((bidx & 31) == l && (bidx >> 5) < 16) lv[bidx >> 5] = -INFINITY;
    if (rr == t) last_idx = bidx;
    int wr = (rr <= t) ? bidx : last_idx;   // rank clamp (pad behaviour)
    if (l == 0) idxbuf[obase + rr] = wr;
  }
}

// ---- trilinear core: one block (4 waves) per (b,h,i) -----------------------
__global__ __launch_bounds__(128) void ho_trilinear(const float* __restrict__ Q,
                                                    const float* __restrict__ K1,
                                                    const float* __restrict__ K2,
                                                    const float* __restrict__ V1,
                                                    const float* __restrict__ V2,
                                                    const int* __restrict__ idxbuf,
                                                    __bf16* __restrict__ O) {
  __shared__ __align__(32) __bf16 sP  [64][64];  // q (.) K1g
  __shared__ __align__(32) __bf16 sK2 [64][64];
  __shared__ __align__(32) __bf16 sV1 [64][64];
  __shared__ __align__(32) __bf16 sV2T[64][64];  // transposed: [d][k]
  __shared__ __align__(32) __bf16 sAl [64][64];  // softmax(A) bf16
  __shared__ float sAcc[64][64];                 // A (f32), later W
  __shared__ float sQ[64];
  __shared__ int   sIdx1[64], sIdx2[64];

  int t = blockIdx.x % TSEQ;
  int h = (blockIdx.x / TSEQ) % NH;
  int b = blockIdx.x / (TSEQ * NH);
  int tid = threadIdx.x;
  size_t rowbase = (size_t)(b * TSEQ + t) * EMB + h * HDIM;

  size_t i1 = (((size_t)0 * BB + b) * NH + h) * TSEQ * KKEEP + (size_t)t * KKEEP;
  size_t i2 = (((size_t)1 * BB + b) * NH + h) * TSEQ * KKEEP + (size_t)t * KKEEP;
  if (tid < 64) { sQ[tid] = Q[rowbase + tid]; sIdx1[tid] = idxbuf[i1 + tid]; }
  else          { sIdx2[tid - 64] = idxbuf[i2 + (tid - 64)]; }
  __syncthreads();

  // gather + convert to bf16
  for (int e = tid; e < 64 * 64; e += 128) {
    int j = e >> 6, d = e & 63;
    int s1 = sIdx1[j], s2 = sIdx2[j];
    size_t r1 = (size_t)(b * TSEQ + s1) * EMB + h * HDIM;
    size_t r2 = (size_t)(b * TSEQ + s2) * EMB + h * HDIM;
    sP  [j][d] = (__bf16)(sQ[d] * K1[r1 + d]);
    sK2 [j][d] = (__bf16)K2[r2 + d];
    sV1 [j][d] = (__bf16)V1[r1 + d];
    sV2T[d][j] = (__bf16)V2[r2 + d];
  }
  __syncthreads();

  const float scale = 0.125f;  // 64^-0.5
  int wave = tid >> 5, l = tid & 31, hi = l >> 4, n = l & 15;
  int row0 = wave * 16;

  // A = P @ K2g^T  (64x64x64): wave owns 16-row strip, 4 col tiles
  {
    v8f acc[4] = {{}, {}, {}, {}};
#pragma unroll
    for (int k0 = 0; k0 < 64; k0 += 32) {
      v16bf a = load_a(&sP[0][0], 64, row0, k0);
#pragma unroll
      for (int tj = 0; tj < 4; ++tj) {
        v16bf bm = load_b(&sK2[0][0], 64, tj * 16, k0);
        acc[tj] = __builtin_amdgcn_wmma_f32_16x16x32_bf16(false, a, false, bm,
                                                          (short)0, acc[tj], false, false);
      }
    }
#pragma unroll
    for (int tj = 0; tj < 4; ++tj)
#pragma unroll
      for (int r = 0; r < 8; ++r)
        sAcc[row0 + r + 8 * hi][tj * 16 + n] = acc[tj][r] * scale;
  }
  __syncthreads();

  // row softmax over k
  if (tid < 64) {
    float m = -INFINITY;
    for (int k = 0; k < 64; ++k) m = fmaxf(m, sAcc[tid][k]);
    float s = 0.0f;
    for (int k = 0; k < 64; ++k) { float ev = __expf(sAcc[tid][k] - m); sAcc[tid][k] = ev; s += ev; }
    float inv = 1.0f / s;
    for (int k = 0; k < 64; ++k) sAl[tid][k] = (__bf16)(sAcc[tid][k] * inv);
  }
  __syncthreads();

  // W = alpha @ V2g  (64x64x64); sV2T is [d][k] so B loads are contiguous in k
  {
    v8f acc[4] = {{}, {}, {}, {}};
#pragma unroll
    for (int k0 = 0; k0 < 64; k0 += 32) {
      v16bf a = load_a(&sAl[0][0], 64, row0, k0);
#pragma unroll
      for (int tj = 0; tj < 4; ++tj) {
        v16bf bm = load_b(&sV2T[0][0], 64, tj * 16, k0);
        acc[tj] = __builtin_amdgcn_wmma_f32_16x16x32_bf16(false, a, false, bm,
                                                          (short)0, acc[tj], false, false);
      }
    }
#pragma unroll
    for (int tj = 0; tj < 4; ++tj)
#pragma unroll
      for (int r = 0; r < 8; ++r)
        sAcc[row0 + r + 8 * hi][tj * 16 + n] = acc[tj][r];
  }
  __syncthreads();

  // out[d] = sum_j V1g[j,d] * W[j,d]
  if (tid < 64) {
    int d = tid;
    float acc = 0.0f;
    for (int j = 0; j < 64; ++j) acc += (float)sV1[j][d] * sAcc[j][d];
    O[rowbase + d] = (__bf16)acc;
  }
}

// ---------------------------------------------------------------------------
extern "C" void kernel_launch(void* const* d_in, const int* in_sizes, int n_in,
                              void* d_out, int out_size, void* d_ws, size_t ws_size,
                              hipStream_t stream) {
  const float* x  = (const float*)d_in[0];   // (2,512,512)
  const float* Wq = (const float*)d_in[1];   // (512,512)
  const float* Wk = (const float*)d_in[2];   // (2,512,512)
  const float* Wv = (const float*)d_in[3];   // (2,512,512)
  const float* Wo = (const float*)d_in[4];   // (512,512)

  const size_t MB = 1u << 20;
  char* ws = (char*)d_ws;
  __bf16* xb   = (__bf16*)(ws + 0);                 // 1 MB  (1024x512 bf16)
  __bf16* WqT  = (__bf16*)(ws + 1 * MB);            // 512 KB each (N x K)
  __bf16* Wk0T = (__bf16*)(ws + 1 * MB + 512 * 1024);
  __bf16* Wk1T = (__bf16*)(ws + 2 * MB);
  __bf16* Wv0T = (__bf16*)(ws + 2 * MB + 512 * 1024);
  __bf16* Wv1T = (__bf16*)(ws + 3 * MB);
  __bf16* WoT  = (__bf16*)(ws + 3 * MB + 512 * 1024);
  float*  Qf   = (float*)(ws + 4 * MB);             // 2 MB each (1024x512 f32)
  float*  K1f  = (float*)(ws + 6 * MB);
  float*  K2f  = (float*)(ws + 8 * MB);
  float*  V1f  = (float*)(ws + 10 * MB);
  float*  V2f  = (float*)(ws + 12 * MB);
  __bf16* Ob   = (__bf16*)(ws + 14 * MB);           // 1 MB (1024x512 bf16)
  int*    idxb = (int*)(ws + 15 * MB);              // 4 MB (2*2*8*512*64 i32)

  // 1) convert x and weights (weights transposed to N x K) to bf16
  {
    int n = MROWS * EMB;
    ho_convert_bf16<<<(n + 255) / 256, 256, 0, stream>>>(x, xb, MROWS, EMB, 0);
    int m = EMB * EMB;
    dim3 g((m + 255) / 256);
    ho_convert_bf16<<<g, 256, 0, stream>>>(Wq,               WqT,  EMB, EMB, 1);
    ho_convert_bf16<<<g, 256, 0, stream>>>(Wk,               Wk0T, EMB, EMB, 1);
    ho_convert_bf16<<<g, 256, 0, stream>>>(Wk + EMB * EMB,   Wk1T, EMB, EMB, 1);
    ho_convert_bf16<<<g, 256, 0, stream>>>(Wv,               Wv0T, EMB, EMB, 1);
    ho_convert_bf16<<<g, 256, 0, stream>>>(Wv + EMB * EMB,   Wv1T, EMB, EMB, 1);
    ho_convert_bf16<<<g, 256, 0, stream>>>(Wo,               WoT,  EMB, EMB, 1);
  }

  // 2) projections via WMMA GEMM
  {
    dim3 grid(MROWS / 64, EMB / 64);
    ho_gemm_bf16<<<grid, 128, 0, stream>>>(xb, WqT,  Qf,  MROWS, EMB, EMB);
    ho_gemm_bf16<<<grid, 128, 0, stream>>>(xb, Wk0T, K1f, MROWS, EMB, EMB);
    ho_gemm_bf16<<<grid, 128, 0, stream>>>(xb, Wk1T, K2f, MROWS, EMB, EMB);
    ho_gemm_bf16<<<grid, 128, 0, stream>>>(xb, Wv0T, V1f, MROWS, EMB, EMB);
    ho_gemm_bf16<<<grid, 128, 0, stream>>>(xb, Wv1T, V2f, MROWS, EMB, EMB);
  }

  // 3) causal top-64 indices for both branches
  {
    dim3 grid((BB * NH * TSEQ) / 4, 2);
    ho_topk<<<grid, 128, 0, stream>>>(Qf, K1f, K2f, idxb);
  }

  // 4) trilinear attention core
  ho_trilinear<<<BB * NH * TSEQ, 128, 0, stream>>>(Qf, K1f, K2f, V1f, V2f, idxb, Ob);

  // 5) final projection -> d_out (f32)
  {
    dim3 grid(MROWS / 64, EMB / 64);
    ho_gemm_bf16<<<grid, 128, 0, stream>>>(Ob, WoT, (float*)d_out, MROWS, EMB, EMB);
  }
}
